// CausalSelfAttention_35476429865094
// MI455X (gfx1250) — matrix operand order repaired and checked by reading, not compiled, not run on hardware
//
#include <hip/hip_runtime.h>
#include <hip/hip_bf16.h>

#define D_MODEL  2048
#define N_HEADS  16
#define HEAD_DIM 128
#define BATCH    2
#define SEQ      2048
#define BT       (BATCH * SEQ)          // 4096 rows
#define QKV_N    (3 * D_MODEL)          // 6144

typedef __attribute__((ext_vector_type(16))) __bf16 v16bf;
typedef __attribute__((ext_vector_type(8)))  float  v8f;

union Frag16 { v16bf v; uint4 q[2]; };

static __device__ __forceinline__ v8f wmma_bf16(const Frag16& a, const Frag16& b, v8f c) {
    // D = A(16x32 bf16) * B(32x16 bf16) + C(16x16 f32)
    return __builtin_amdgcn_wmma_f32_16x16x32_bf16(false, a.v, false, b.v,
                                                   (short)0, c, false, false);
}

static __device__ __forceinline__ v8f vzero8() {
    v8f z;
#pragma unroll
    for (int r = 0; r < 8; ++r) z[r] = 0.0f;
    return z;
}

// Async global->LDS copy, 16 bytes per lane (ASYNCcnt-tracked, CDNA5).
// lds_off: LDS byte offset (low 32 bits of generic pointer), gaddr: global addr.
static __device__ __forceinline__ void async_copy_b128(unsigned lds_off, const void* gaddr) {
    asm volatile("global_load_async_to_lds_b128 %0, %1, off"
                 :: "v"(lds_off), "v"((unsigned long long)(uintptr_t)gaddr)
                 : "memory");
}

// ---------------------------------------------------------------------------
// 1) float -> bf16 elementwise convert
// ---------------------------------------------------------------------------
__global__ void cvt_f32_bf16(const float* __restrict__ in, __bf16* __restrict__ out, int n) {
    int i = blockIdx.x * 256 + threadIdx.x;
    if (i < n) out[i] = (__bf16)in[i];
}

// ---------------------------------------------------------------------------
// 2) transpose + convert: W[K,N] f32 -> Wt[N,K] bf16  (so WMMA B-fragments are
//    contiguous along K)
// ---------------------------------------------------------------------------
__global__ void cvt_transpose(const float* __restrict__ in, __bf16* __restrict__ out,
                              int K, int N) {
    int i = blockIdx.x * 256 + threadIdx.x;
    if (i >= K * N) return;
    int k = i / N;
    int n = i - k * N;
    out[(size_t)n * K + k] = (__bf16)in[i];
}

// ---------------------------------------------------------------------------
// 3) bf16 GEMM via WMMA:  C[M,N](f32) = A[M,K] * Bt[N,K]^T
//    4-wave workgroup computes a 64x64 tile. The 64x32 B sub-tile is staged in
//    LDS via double-buffered global_load_async_to_lds_b128 and shared by all
//    4 waves; A fragments stream from global (distinct rows per wave).
// ---------------------------------------------------------------------------
__global__ void gemm_bf16_tiled(const __bf16* __restrict__ A, const __bf16* __restrict__ Bt,
                                float* __restrict__ C, int M, int N, int K) {
    __shared__ __attribute__((aligned(16))) __bf16 ldsb[2][64 * 32];   // 2 x 4KB

    const int tid  = threadIdx.x;        // 0..127
    const int lane = tid & 31;
    const int w    = tid >> 5;           // wave id 0..3
    const int lo = lane & 15, hi = lane >> 4;
    const int m0 = blockIdx.y * 64 + w * 16;
    const int n0 = blockIdx.x * 64;

    v8f acc[4];
#pragma unroll
    for (int j = 0; j < 4; ++j) acc[j] = vzero8();

    const __bf16* arow = A + (size_t)(m0 + lo) * K;

    // async slice for this thread: two 16B chunks of the 64x32 tile
    const int l0 = tid, l1 = tid + 128;           // 0..255
    const int r0 = l0 >> 2, c0 = (l0 & 3) * 8;    // row 0..63, elem col 0/8/16/24
    const int r1 = l1 >> 2, c1 = (l1 & 3) * 8;

    const int nkt = K / 32;
    // prefetch tile 0
    async_copy_b128((unsigned)(uintptr_t)&ldsb[0][r0 * 32 + c0],
                    Bt + (size_t)(n0 + r0) * K + c0);
    async_copy_b128((unsigned)(uintptr_t)&ldsb[0][r1 * 32 + c1],
                    Bt + (size_t)(n0 + r1) * K + c1);

    for (int kt = 0; kt < nkt; ++kt) {
        const int kk  = kt * 32;
        const int cur = kt & 1;
        if (kt + 1 < nkt) {
            const int nxt = cur ^ 1;
            const int kk2 = kk + 32;
            async_copy_b128((unsigned)(uintptr_t)&ldsb[nxt][r0 * 32 + c0],
                            Bt + (size_t)(n0 + r0) * K + kk2 + c0);
            async_copy_b128((unsigned)(uintptr_t)&ldsb[nxt][r1 * 32 + c1],
                            Bt + (size_t)(n0 + r1) * K + kk2 + c1);
            asm volatile("s_wait_asynccnt 0x2" ::: "memory");  // current tile landed
        } else {
            asm volatile("s_wait_asynccnt 0x0" ::: "memory");
        }
        __syncthreads();

        Frag16 a;   // A-layout: K = hi*8+e (e<8), 16+hi*8+(e-8) (e>=8)
        a.q[0] = *reinterpret_cast<const uint4*>(arow + kk + hi * 8);
        a.q[1] = *reinterpret_cast<const uint4*>(arow + kk + 16 + hi * 8);
#pragma unroll
        for (int j = 0; j < 4; ++j) {
            Frag16 b;   // B-layout: lane=col n, K = hi*8+e (e<8), 16+hi*8+(e-8) (e>=8)
            const __bf16* bp = &ldsb[cur][(j * 16 + lo) * 32];
            b.q[0] = *reinterpret_cast<const uint4*>(bp + hi * 8);
            b.q[1] = *reinterpret_cast<const uint4*>(bp + 16 + hi * 8);
            acc[j] = wmma_bf16(a, b, acc[j]);
        }
        __syncthreads();   // protect buf `cur` before it is overwritten at kt+2
    }
#pragma unroll
    for (int j = 0; j < 4; ++j)
#pragma unroll
        for (int r = 0; r < 8; ++r) {
            int m = r + 8 * hi;                       // C/D layout: VGPR r, half hi
            C[(size_t)(m0 + m) * N + n0 + j * 16 + lo] = acc[j][r];
        }
}

// ---------------------------------------------------------------------------
// 4) RoPE + layout: qkv f32 [B,T,3D] ->
//      qh,kh bf16 [B,H,T,hd] (RoPE applied), vT bf16 [B,H,hd,T]
// ---------------------------------------------------------------------------
__global__ void rope_layout(const float* __restrict__ qkv, __bf16* __restrict__ qh,
                            __bf16* __restrict__ kh, __bf16* __restrict__ vT) {
    int idx = blockIdx.x * 256 + threadIdx.x;     // B*T*H*hd = 2^23 threads
    int d = idx & (HEAD_DIM - 1);
    int h = (idx >> 7) & (N_HEADS - 1);
    int t = (idx >> 11) & (SEQ - 1);
    int b = idx >> 22;

    const size_t rowbase = (size_t)(b * SEQ + t) * QKV_N;
    const int col = h * HEAD_DIM + d;
    float q = qkv[rowbase + col];
    float k = qkv[rowbase + D_MODEL + col];
    float v = qkv[rowbase + 2 * D_MODEL + col];

    // inv_freq[i] = 10000^(-i/64);  ln(10000)/64 = 0.14391156831212787
    int dh = d & 63;
    float freq = __expf(-(float)dh * 0.14391156831212787f);
    float ang = (float)t * freq;
    float s, c;
    __sincosf(ang, &s, &c);

    int d2 = (d < 64) ? d + 64 : d - 64;
    float qp = qkv[rowbase + h * HEAD_DIM + d2];
    float kp = qkv[rowbase + D_MODEL + h * HEAD_DIM + d2];
    float qrot = (d < 64) ? -qp : qp;
    float krot = (d < 64) ? -kp : kp;

    size_t o = ((size_t)(b * N_HEADS + h) * SEQ + t) * HEAD_DIM + d;
    qh[o] = (__bf16)(q * c + qrot * s);
    kh[o] = (__bf16)(k * c + krot * s);
    vT[((size_t)(b * N_HEADS + h) * HEAD_DIM + d) * SEQ + t] = (__bf16)v;
}

// ---------------------------------------------------------------------------
// 5) Flash-attention: one wave per (b,h, 16-query tile). Key tiles of 32.
//    Scores via 4x wmma over hd, online softmax (causal + ALiBi), P staged in
//    LDS (D-layout -> A-layout), PV via 8x wmma across hd. Output bf16 [B,T,D].
// ---------------------------------------------------------------------------
__global__ void attn_kernel(const __bf16* __restrict__ qh, const __bf16* __restrict__ kh,
                            const __bf16* __restrict__ vT, __bf16* __restrict__ attn) {
    const int lane = threadIdx.x;
    const int lo = lane & 15, hi = lane >> 4;
    const int qt = blockIdx.x;
    const int bh = blockIdx.y;
    const int h = bh & (N_HEADS - 1);
    const int b = bh >> 4;
    const int q0 = qt * 16;

    const __bf16* qbase = qh + (size_t)bh * SEQ * HEAD_DIM;
    const __bf16* kbase = kh + (size_t)bh * SEQ * HEAD_DIM;
    const __bf16* vbase = vT + (size_t)bh * HEAD_DIM * SEQ;

    __shared__ __attribute__((aligned(16))) __bf16 plds[16 * 32];

    // Q fragments for this 16-row tile (hd = 4 chunks of K=32)
    Frag16 qf[4];
    const __bf16* qrow = qbase + (size_t)(q0 + lo) * HEAD_DIM;
#pragma unroll
    for (int cc = 0; cc < 4; ++cc) {
        qf[cc].q[0] = *reinterpret_cast<const uint4*>(qrow + cc * 32 + hi * 8);
        qf[cc].q[1] = *reinterpret_cast<const uint4*>(qrow + cc * 32 + 16 + hi * 8);
    }

    v8f o[8];
#pragma unroll
    for (int tt = 0; tt < 8; ++tt) o[tt] = vzero8();
    float rmax[8], rsum[8];
#pragma unroll
    for (int r = 0; r < 8; ++r) { rmax[r] = -__builtin_inff(); rsum[r] = 0.0f; }

    const float slope = exp2f(-0.5f * (float)(h + 1));   // 2^(-8*(h+1)/16)
    const float sc = 0.08838834764831845f;               // 1/sqrt(128)

    const int nkt = (q0 + 15) / 32 + 1;
    for (int kt = 0; kt < nkt; ++kt) {
        const int k0 = kt * 32;
        v8f s0 = vzero8(), s1 = vzero8();
#pragma unroll
        for (int cc = 0; cc < 4; ++cc) {
            Frag16 b0, b1;   // B-frag: lane=key col, K=hd dim hi*8+e (e<8), 16+hi*8+(e-8) (e>=8)
            const __bf16* kp0 = kbase + (size_t)(k0 + lo) * HEAD_DIM + cc * 32 + hi * 8;
            const __bf16* kp1 = kbase + (size_t)(k0 + 16 + lo) * HEAD_DIM + cc * 32 + hi * 8;
            b0.q[0] = *reinterpret_cast<const uint4*>(kp0);
            b0.q[1] = *reinterpret_cast<const uint4*>(kp0 + 16);
            b1.q[0] = *reinterpret_cast<const uint4*>(kp1);
            b1.q[1] = *reinterpret_cast<const uint4*>(kp1 + 16);
            s0 = wmma_bf16(qf[cc], b0, s0);
            s1 = wmma_bf16(qf[cc], b1, s1);
        }
        // scale + causal mask + ALiBi, per-element (row m = r+8*hi, col = lane lo)
        float tmax[8];
#pragma unroll
        for (int r = 0; r < 8; ++r) {
            int i = q0 + r + 8 * hi;
            int j0 = k0 + lo, j1 = k0 + 16 + lo;
            float e0 = (j0 <= i) ? s0[r] * sc + slope * (float)(j0 - i) : -__builtin_inff();
            float e1 = (j1 <= i) ? s1[r] * sc + slope * (float)(j1 - i) : -__builtin_inff();
            s0[r] = e0; s1[r] = e1;
            tmax[r] = fmaxf(e0, e1);
        }
#pragma unroll
        for (int r = 0; r < 8; ++r) {
            tmax[r] = fmaxf(tmax[r], __shfl_xor(tmax[r], 1, 32));
            tmax[r] = fmaxf(tmax[r], __shfl_xor(tmax[r], 2, 32));
            tmax[r] = fmaxf(tmax[r], __shfl_xor(tmax[r], 4, 32));
            tmax[r] = fmaxf(tmax[r], __shfl_xor(tmax[r], 8, 32));
        }
        float corr[8], tsum[8];
#pragma unroll
        for (int r = 0; r < 8; ++r) {
            float nm = fmaxf(rmax[r], tmax[r]);
            corr[r] = __expf(rmax[r] - nm);
            rmax[r] = nm;
            float p0 = __expf(s0[r] - nm);
            float p1 = __expf(s1[r] - nm);
            s0[r] = p0; s1[r] = p1;
            tsum[r] = p0 + p1;
        }
#pragma unroll
        for (int r = 0; r < 8; ++r) {
            tsum[r] += __shfl_xor(tsum[r], 1, 32);
            tsum[r] += __shfl_xor(tsum[r], 2, 32);
            tsum[r] += __shfl_xor(tsum[r], 4, 32);
            tsum[r] += __shfl_xor(tsum[r], 8, 32);
            rsum[r] = rsum[r] * corr[r] + tsum[r];
        }
#pragma unroll
        for (int tt = 0; tt < 8; ++tt)
#pragma unroll
            for (int r = 0; r < 8; ++r) o[tt][r] *= corr[r];

        // stage P (16x32 bf16) through LDS: C/D layout -> A layout
#pragma unroll
        for (int r = 0; r < 8; ++r) {
            int m = r + 8 * hi;
            plds[m * 32 + lo]      = (__bf16)s0[r];
            plds[m * 32 + 16 + lo] = (__bf16)s1[r];
        }
        __syncthreads();
        Frag16 pf;
        pf.q[0] = *reinterpret_cast<const uint4*>(&plds[lo * 32 + hi * 8]);
        pf.q[1] = *reinterpret_cast<const uint4*>(&plds[lo * 32 + 16 + hi * 8]);
        __syncthreads();

        // PV: P(16x32) * V(32keys x 16dv), V from transposed layout (contiguous keys)
#pragma unroll
        for (int tt = 0; tt < 8; ++tt) {
            Frag16 vf;
            const __bf16* vp = vbase + (size_t)(tt * 16 + lo) * SEQ + k0 + hi * 8;
            vf.q[0] = *reinterpret_cast<const uint4*>(vp);
            vf.q[1] = *reinterpret_cast<const uint4*>(vp + 16);
            o[tt] = wmma_bf16(pf, vf, o[tt]);
        }
    }

    // epilogue: normalize and scatter to [B,T,D] bf16
#pragma unroll
    for (int tt = 0; tt < 8; ++tt)
#pragma unroll
        for (int r = 0; r < 8; ++r) {
            int m = r + 8 * hi;
            float val = o[tt][r] / rsum[r];
            attn[(size_t)(b * SEQ + q0 + m) * D_MODEL + h * HEAD_DIM + tt * 16 + lo] =
                (__bf16)val;
        }
}

// ---------------------------------------------------------------------------
extern "C" void kernel_launch(void* const* d_in, const int* in_sizes, int n_in,
                              void* d_out, int out_size, void* d_ws, size_t ws_size,
                              hipStream_t stream) {
    const float* x      = (const float*)d_in[0];
    // d_in[1] = causal_mask (bool) — causality computed analytically, unused
    const float* W_qkv  = (const float*)d_in[2];
    const float* W_proj = (const float*)d_in[3];
    float* y = (float*)d_out;

    // workspace carve-up (256B aligned)
    char* ws = (char*)d_ws;
    size_t off = 0;
    auto take = [&](size_t bytes) -> void* {
        void* p = ws + off;
        off += (bytes + 255) & ~(size_t)255;
        return p;
    };
    __bf16* xh     = (__bf16*)take((size_t)BT * D_MODEL * 2);
    __bf16* wqkvT  = (__bf16*)take((size_t)D_MODEL * QKV_N * 2);
    __bf16* wprojT = (__bf16*)take((size_t)D_MODEL * D_MODEL * 2);
    __bf16* qh     = (__bf16*)take((size_t)BT * D_MODEL * 2);
    __bf16* kh     = (__bf16*)take((size_t)BT * D_MODEL * 2);
    __bf16* vT     = (__bf16*)take((size_t)BT * D_MODEL * 2);
    float*  qkv    = (float*)take((size_t)BT * QKV_N * 4);
    __bf16* attn_h = (__bf16*)qkv;   // alias: qkv f32 dead after rope_layout

    // 1) convert x
    {
        int n = BT * D_MODEL;
        cvt_f32_bf16<<<dim3((n + 255) / 256), dim3(256), 0, stream>>>(x, xh, n);
    }
    // 2) transpose+convert weights
    {
        int n = D_MODEL * QKV_N;
        cvt_transpose<<<dim3((n + 255) / 256), dim3(256), 0, stream>>>(W_qkv, wqkvT, D_MODEL, QKV_N);
        int m = D_MODEL * D_MODEL;
        cvt_transpose<<<dim3((m + 255) / 256), dim3(256), 0, stream>>>(W_proj, wprojT, D_MODEL, D_MODEL);
    }
    // 3) qkv = x @ W_qkv  (f32 out) — 64x64 tiles, 4 waves/block
    gemm_bf16_tiled<<<dim3(QKV_N / 64, BT / 64), dim3(128), 0, stream>>>(
        xh, wqkvT, qkv, BT, QKV_N, D_MODEL);
    // 4) RoPE + relayout
    {
        int n = BT * D_MODEL;
        rope_layout<<<dim3(n / 256), dim3(256), 0, stream>>>(qkv, qh, kh, vT);
    }
    // 5) attention (one wave per 16-query tile per head)
    attn_kernel<<<dim3(SEQ / 16, BATCH * N_HEADS), dim3(32), 0, stream>>>(qh, kh, vT, attn_h);
    // 6) y = attn @ W_proj  (f32 out to d_out)
    gemm_bf16_tiled<<<dim3(D_MODEL / 64, BT / 64), dim3(128), 0, stream>>>(
        attn_h, wprojT, y, BT, D_MODEL, D_MODEL);
    (void)in_sizes; (void)n_in; (void)out_size; (void)ws_size;
}
